// MultiHeadAttention_28406913696252
// MI455X (gfx1250) — matrix operand order repaired and checked by reading, not compiled, not run on hardware
//
#include <hip/hip_runtime.h>
#include <stdint.h>

#define B_  2
#define S_  2048
#define F_  1024
#define H_  16
#define DK_ 64
#define DM_ 1024
#define HD_ (H_ * DK_)   // 1024

typedef __attribute__((ext_vector_type(16))) __bf16 v16bf;
typedef __attribute__((ext_vector_type(8)))  float  v8f;

union FragU { v16bf v; uint32_t u[8]; };

#define WMMA_BF16(a, b, c) \
  __builtin_amdgcn_wmma_f32_16x16x32_bf16(false, (a), false, (b), (short)0, (c), false, false)

__device__ __forceinline__ uint16_t f2bf(float x) {
  uint32_t u = __builtin_bit_cast(uint32_t, x);
  u += 0x7FFFu + ((u >> 16) & 1u);         // round-to-nearest-even
  return (uint16_t)(u >> 16);
}
__device__ __forceinline__ uint32_t pack2bf(float lo, float hi) {
  return (uint32_t)f2bf(lo) | ((uint32_t)f2bf(hi) << 16);
}
__device__ __forceinline__ v8f vzero8() {
  v8f z;
#pragma unroll
  for (int i = 0; i < 8; ++i) z[i] = 0.0f;
  return z;
}

// A fragment, 16x32 bf16 (MxK), K-contiguous source rows.
// ISA layout: lane m = lane&15; half=lane>>4; VGPR0-3: K = half*8 + 2v..;
// VGPR4-7: K = 16 + half*8 + 2v..
__device__ __forceinline__ v16bf load_a_frag(const uint16_t* __restrict__ base,
                                             int ldk, int lane) {
  const int half = (lane >> 4) & 1;
  const uint16_t* row = base + (size_t)(lane & 15) * ldk;
  FragU f;
#pragma unroll
  for (int v = 0; v < 4; ++v)
    f.u[v] = *(const uint32_t*)(row + half * 8 + 2 * v);
#pragma unroll
  for (int v = 0; v < 4; ++v)
    f.u[4 + v] = *(const uint32_t*)(row + 16 + half * 8 + 2 * v);
  return f.v;
}

// B fragment, 32x16 bf16 (KxN), loaded from B^T storage (N rows x K cols).
// ISA layout: lane n = lane&15; half=lane>>4; VGPR0-3: K = half*8 + 2v..;
// VGPR4-7: K = 16 + half*8 + 2v..
__device__ __forceinline__ v16bf load_b_frag(const uint16_t* __restrict__ baseT,
                                             int ldk, int lane) {
  const int half = (lane >> 4) & 1;
  const uint16_t* row = baseT + (size_t)(lane & 15) * ldk;
  FragU f;
#pragma unroll
  for (int v = 0; v < 4; ++v)
    f.u[v] = *(const uint32_t*)(row + half * 8 + 2 * v);
#pragma unroll
  for (int v = 0; v < 4; ++v)
    f.u[4 + v] = *(const uint32_t*)(row + 16 + half * 8 + 2 * v);
  return f.v;
}

__device__ __forceinline__ float redmax16(float x) {
#pragma unroll
  for (int m = 8; m >= 1; m >>= 1) x = fmaxf(x, __shfl_xor(x, m, 32));
  return x;
}
__device__ __forceinline__ float redsum16(float x) {
#pragma unroll
  for (int m = 8; m >= 1; m >>= 1) x += __shfl_xor(x, m, 32);
  return x;
}

// ---------------- conversion kernels (fp32 -> bf16, with transposes) -------

__global__ __launch_bounds__(256) void cvt_x_kernel(const float* __restrict__ x,
                                                    uint16_t* __restrict__ xb) {
  size_t i = ((size_t)blockIdx.x * blockDim.x + threadIdx.x) * 4;
  const float4 f = *(const float4*)(x + i);
  uint32_t* o = (uint32_t*)(xb + i);
  o[0] = pack2bf(f.x, f.y);
  o[1] = pack2bf(f.z, f.w);
}

// [H,F,DK] -> bf16 [H,DK,F]
__global__ __launch_bounds__(256) void cvt_wT_kernel(const float* __restrict__ w,
                                                     uint16_t* __restrict__ wt) {
  int o = blockIdx.x * blockDim.x + threadIdx.x;   // H*DK*F
  int f = o & (F_ - 1);
  int d = (o >> 10) & (DK_ - 1);
  int h = o >> 16;
  wt[o] = f2bf(w[(size_t)(h * F_ + f) * DK_ + d]);
}

// [HD, DM] -> bf16 [DM, HD]
__global__ __launch_bounds__(256) void cvt_woT_kernel(const float* __restrict__ wo,
                                                      uint16_t* __restrict__ wt) {
  int o = blockIdx.x * blockDim.x + threadIdx.x;   // DM*HD
  int k = o & (HD_ - 1);
  int n = o >> 10;
  wt[o] = f2bf(wo[(size_t)k * DM_ + n]);
}

// ---------------- QKV projection: per-wave 32x64 tile, K-loop over F -------
// wid -> {mat(q/k/v), bh, sblk}; Q,K stored [BH,S,DK]; V stored transposed [BH,DK,S]
__global__ __launch_bounds__(256) void qkv_kernel(const uint16_t* __restrict__ Xb,
                                                  const uint16_t* __restrict__ WqT,
                                                  const uint16_t* __restrict__ WkT,
                                                  const uint16_t* __restrict__ WvT,
                                                  uint16_t* __restrict__ Qb,
                                                  uint16_t* __restrict__ Kb,
                                                  uint16_t* __restrict__ Vt) {
  const int tid = threadIdx.x, lane = tid & 31;
  const int wid = blockIdx.x * 8 + (tid >> 5);     // 6144 waves
  const int sblk = wid & 63;                       // 64 row-tiles of 32
  const int tmp  = wid >> 6;
  const int bh   = tmp & 31;
  const int mat  = tmp >> 5;                       // 0=q 1=k 2=v
  const int b = bh >> 4, h = bh & 15;

  const uint16_t* arow = Xb + ((size_t)b * S_ + sblk * 32) * F_;
  const uint16_t* wt = (mat == 0 ? WqT : (mat == 1 ? WkT : WvT)) + (size_t)h * DK_ * F_;

  v8f acc[2][4];
#pragma unroll
  for (int r = 0; r < 2; ++r)
#pragma unroll
    for (int f = 0; f < 4; ++f) acc[r][f] = vzero8();

  for (int k0 = 0; k0 < F_; k0 += 32) {
    __builtin_prefetch(arow + k0 + 128, 0, 3);
    // batch ALL fragment loads before any WMMA -> staged s_wait_loadcnt
    v16bf a0 = load_a_frag(arow + k0,            F_, lane);
    v16bf a1 = load_a_frag(arow + 16 * F_ + k0,  F_, lane);
    v16bf b0 = load_b_frag(wt + 0 * 16 * F_ + k0, F_, lane);
    v16bf b1 = load_b_frag(wt + 1 * 16 * F_ + k0, F_, lane);
    v16bf b2 = load_b_frag(wt + 2 * 16 * F_ + k0, F_, lane);
    v16bf b3 = load_b_frag(wt + 3 * 16 * F_ + k0, F_, lane);
    acc[0][0] = WMMA_BF16(a0, b0, acc[0][0]);
    acc[1][0] = WMMA_BF16(a1, b0, acc[1][0]);
    acc[0][1] = WMMA_BF16(a0, b1, acc[0][1]);
    acc[1][1] = WMMA_BF16(a1, b1, acc[1][1]);
    acc[0][2] = WMMA_BF16(a0, b2, acc[0][2]);
    acc[1][2] = WMMA_BF16(a1, b2, acc[1][2]);
    acc[0][3] = WMMA_BF16(a0, b3, acc[0][3]);
    acc[1][3] = WMMA_BF16(a1, b3, acc[1][3]);
  }

  const int half = lane >> 4, nr = lane & 15;
  if (mat < 2) {
    uint16_t* dst = ((mat == 0) ? Qb : Kb) + ((size_t)bh * S_ + sblk * 32) * DK_;
#pragma unroll
    for (int r = 0; r < 2; ++r)
#pragma unroll
      for (int v = 0; v < 8; ++v) {
        uint16_t* p = dst + (size_t)(r * 16 + half * 8 + v) * DK_ + nr;
        p[0]  = f2bf(acc[r][0][v]);
        p[16] = f2bf(acc[r][1][v]);
        p[32] = f2bf(acc[r][2][v]);
        p[48] = f2bf(acc[r][3][v]);
      }
  } else {
    uint16_t* dst = Vt + (size_t)bh * DK_ * S_ + (size_t)nr * S_ + sblk * 32;
#pragma unroll
    for (int r = 0; r < 2; ++r)
#pragma unroll
      for (int v = 0; v < 8; ++v) {
        int srow = r * 16 + half * 8 + v;
        dst[(size_t)0 * 16 * S_ + srow] = f2bf(acc[r][0][v]);
        dst[(size_t)1 * 16 * S_ + srow] = f2bf(acc[r][1][v]);
        dst[(size_t)2 * 16 * S_ + srow] = f2bf(acc[r][2][v]);
        dst[(size_t)3 * 16 * S_ + srow] = f2bf(acc[r][3][v]);
      }
  }
}

// ---------------- flash attention: one wave = 16 query rows ----------------
__global__ __launch_bounds__(256) void attn_kernel(const uint16_t* __restrict__ Qb,
                                                   const uint16_t* __restrict__ Kb,
                                                   const uint16_t* __restrict__ Vt,
                                                   uint16_t* __restrict__ feats) {
  __shared__ __align__(16) uint16_t pbuf[8][16 * 32];   // per-wave P tile (8 KB)

  const int tid = threadIdx.x, lane = tid & 31, wslot = tid >> 5;
  const int wid = blockIdx.x * 8 + wslot;               // 4096 waves
  const int sblk = wid & 127;
  const int bh   = wid >> 7;
  const int b = bh >> 4, h = bh & 15;
  const int sq0 = sblk * 16;
  const int half = lane >> 4, nr = lane & 15;

  const uint16_t* qbase = Qb + ((size_t)bh * S_ + sq0) * DK_;
  const uint16_t* kbase = Kb + (size_t)bh * S_ * DK_;
  const uint16_t* vbase = Vt + (size_t)bh * DK_ * S_;
  uint16_t* pw = &pbuf[wslot][0];

  const v16bf qa0 = load_a_frag(qbase + 0,  DK_, lane);
  const v16bf qa1 = load_a_frag(qbase + 32, DK_, lane);

  v8f o0 = vzero8(), o1 = vzero8(), o2 = vzero8(), o3 = vzero8();
  float rmax[8], rsum[8];
#pragma unroll
  for (int v = 0; v < 8; ++v) { rmax[v] = -3.0e38f; rsum[v] = 0.0f; }

  const float scale = 0.125f;                           // 1/sqrt(64)

  for (int t0 = 0; t0 < S_; t0 += 32) {
    // Issue ALL global fragment loads for this key tile up front; the V
    // fragments then land while score WMMAs + softmax VALU work executes.
    const uint16_t* kt0 = kbase + (size_t)t0 * DK_;
    const uint16_t* kt1 = kbase + (size_t)(t0 + 16) * DK_;
    const uint16_t* vt  = vbase + t0;
    __builtin_prefetch(kt0 + 32 * DK_, 0, 3);
    v16bf kb00 = load_b_frag(kt0,      DK_, lane);
    v16bf kb01 = load_b_frag(kt0 + 32, DK_, lane);
    v16bf kb10 = load_b_frag(kt1,      DK_, lane);
    v16bf kb11 = load_b_frag(kt1 + 32, DK_, lane);
    v16bf vb0 = load_b_frag(vt + (size_t)0 * 16 * S_, S_, lane);
    v16bf vb1 = load_b_frag(vt + (size_t)1 * 16 * S_, S_, lane);
    v16bf vb2 = load_b_frag(vt + (size_t)2 * 16 * S_, S_, lane);
    v16bf vb3 = load_b_frag(vt + (size_t)3 * 16 * S_, S_, lane);

    // S = Q K^T for two 16-key column tiles
    v8f sc0 = vzero8(), sc1 = vzero8();
    sc0 = WMMA_BF16(qa0, kb00, sc0);
    sc0 = WMMA_BF16(qa1, kb01, sc0);
    sc1 = WMMA_BF16(qa0, kb10, sc1);
    sc1 = WMMA_BF16(qa1, kb11, sc1);

    // streaming softmax update (per C-fragment row v)
    float p0[8], p1[8];
#pragma unroll
    for (int v = 0; v < 8; ++v) {
      float a = sc0[v] * scale;
      float c = sc1[v] * scale;
      float mx = redmax16(fmaxf(a, c));
      float mnew = fmaxf(rmax[v], mx);
      float corr = __expf(rmax[v] - mnew);
      rmax[v] = mnew;
      float e0 = __expf(a - mnew);
      float e1 = __expf(c - mnew);
      p0[v] = e0; p1[v] = e1;
      rsum[v] = rsum[v] * corr + redsum16(e0 + e1);
      o0[v] *= corr; o1[v] *= corr; o2[v] *= corr; o3[v] *= corr;
    }
    // P (C-frag layout, f32) -> LDS bf16 16x32 tile -> A fragment (same wave:
    // DS ops are in-order, no barrier needed)
#pragma unroll
    for (int v = 0; v < 8; ++v) {
      int row = half * 8 + v;
      pw[row * 32 + nr]      = f2bf(p0[v]);
      pw[row * 32 + 16 + nr] = f2bf(p1[v]);
    }
    v16bf pa = load_a_frag(pw, 32, lane);
    // O += P V   (V fragments already resident)
    o0 = WMMA_BF16(pa, vb0, o0);
    o1 = WMMA_BF16(pa, vb1, o1);
    o2 = WMMA_BF16(pa, vb2, o2);
    o3 = WMMA_BF16(pa, vb3, o3);
  }

  // normalize and store feats [B,S,H*DK] bf16
  uint16_t* dst = feats + ((size_t)b * S_ + sq0 + half * 8) * HD_ + h * DK_ + nr;
#pragma unroll
  for (int v = 0; v < 8; ++v) {
    float inv = 1.0f / rsum[v];
    uint16_t* r = dst + (size_t)v * HD_;
    r[0]  = f2bf(o0[v] * inv);
    r[16] = f2bf(o1[v] * inv);
    r[32] = f2bf(o2[v] * inv);
    r[48] = f2bf(o3[v] * inv);
  }
}

// ---------------- output projection: feats[4096x1024] @ Wo -> f32 ----------
__global__ __launch_bounds__(256) void oproj_kernel(const uint16_t* __restrict__ feats,
                                                    const uint16_t* __restrict__ WoT,
                                                    float* __restrict__ out) {
  const int tid = threadIdx.x, lane = tid & 31;
  const int wid = blockIdx.x * 8 + (tid >> 5);   // 2048 waves
  const int nblk = wid & 15;                     // 16 tiles of 64 cols
  const int mblk = wid >> 4;                     // 128 tiles of 32 rows

  const uint16_t* arow = feats + (size_t)mblk * 32 * HD_;
  const uint16_t* bt   = WoT + (size_t)nblk * 64 * HD_;

  v8f acc[2][4];
#pragma unroll
  for (int r = 0; r < 2; ++r)
#pragma unroll
    for (int f = 0; f < 4; ++f) acc[r][f] = vzero8();

  for (int k0 = 0; k0 < HD_; k0 += 32) {
    __builtin_prefetch(arow + k0 + 128, 0, 3);
    v16bf a0 = load_a_frag(arow + k0,             HD_, lane);
    v16bf a1 = load_a_frag(arow + 16 * HD_ + k0,  HD_, lane);
    v16bf b0 = load_b_frag(bt + 0 * 16 * HD_ + k0, HD_, lane);
    v16bf b1 = load_b_frag(bt + 1 * 16 * HD_ + k0, HD_, lane);
    v16bf b2 = load_b_frag(bt + 2 * 16 * HD_ + k0, HD_, lane);
    v16bf b3 = load_b_frag(bt + 3 * 16 * HD_ + k0, HD_, lane);
    acc[0][0] = WMMA_BF16(a0, b0, acc[0][0]);
    acc[1][0] = WMMA_BF16(a1, b0, acc[1][0]);
    acc[0][1] = WMMA_BF16(a0, b1, acc[0][1]);
    acc[1][1] = WMMA_BF16(a1, b1, acc[1][1]);
    acc[0][2] = WMMA_BF16(a0, b2, acc[0][2]);
    acc[1][2] = WMMA_BF16(a1, b2, acc[1][2]);
    acc[0][3] = WMMA_BF16(a0, b3, acc[0][3]);
    acc[1][3] = WMMA_BF16(a1, b3, acc[1][3]);
  }

  const int half = lane >> 4, nr = lane & 15;
#pragma unroll
  for (int r = 0; r < 2; ++r) {
    float* orow = out + (size_t)(mblk * 32 + r * 16 + half * 8) * DM_ + nblk * 64 + nr;
#pragma unroll
    for (int v = 0; v < 8; ++v) {
      float* p = orow + (size_t)v * DM_;
      __builtin_nontemporal_store(acc[r][0][v], p + 0);
      __builtin_nontemporal_store(acc[r][1][v], p + 16);
      __builtin_nontemporal_store(acc[r][2][v], p + 32);
      __builtin_nontemporal_store(acc[r][3][v], p + 48);
    }
  }
}

extern "C" void kernel_launch(void* const* d_in, const int* in_sizes, int n_in,
                              void* d_out, int out_size, void* d_ws, size_t ws_size,
                              hipStream_t stream) {
  (void)in_sizes; (void)n_in; (void)out_size; (void)ws_size;
  const float* x  = (const float*)d_in[0];
  const float* wq = (const float*)d_in[1];
  const float* wk = (const float*)d_in[2];
  const float* wv = (const float*)d_in[3];
  const float* wo = (const float*)d_in[4];
  float* out = (float*)d_out;

  const size_t N_X  = (size_t)B_ * S_ * F_;        // 4 Mi
  const size_t N_W  = (size_t)H_ * F_ * DK_;       // 1 Mi
  const size_t N_WO = (size_t)DM_ * HD_;           // 1 Mi
  const size_t N_QK = (size_t)B_ * H_ * S_ * DK_;  // 4 Mi

  uint16_t* ws  = (uint16_t*)d_ws;                 // 48 MB total bf16 scratch
  uint16_t* Xb  = ws;
  uint16_t* WqT = Xb  + N_X;
  uint16_t* WkT = WqT + N_W;
  uint16_t* WvT = WkT + N_W;
  uint16_t* WoT = WvT + N_W;
  uint16_t* Qb  = WoT + N_WO;
  uint16_t* Kb  = Qb  + N_QK;
  uint16_t* Vt  = Kb  + N_QK;
  uint16_t* Ft  = Vt  + N_QK;

  cvt_x_kernel  <<<(unsigned)(N_X / (256 * 4)), 256, 0, stream>>>(x, Xb);
  cvt_wT_kernel <<<(unsigned)(N_W / 256),       256, 0, stream>>>(wq, WqT);
  cvt_wT_kernel <<<(unsigned)(N_W / 256),       256, 0, stream>>>(wk, WkT);
  cvt_wT_kernel <<<(unsigned)(N_W / 256),       256, 0, stream>>>(wv, WvT);
  cvt_woT_kernel<<<(unsigned)(N_WO / 256),      256, 0, stream>>>(wo, WoT);

  qkv_kernel  <<<768, 256, 0, stream>>>(Xb, WqT, WkT, WvT, Qb, Kb, Vt);
  attn_kernel <<<512, 256, 0, stream>>>(Qb, Kb, Vt, Ft);
  oproj_kernel<<<256, 256, 0, stream>>>(Ft, WoT, out);
}
